// CrossAttention_84310208020733
// MI455X (gfx1250) — matrix operand order repaired and checked by reading, not compiled, not run on hardware
//
#include <hip/hip_runtime.h>
#include <hip/hip_fp16.h>

// CDNA5 (gfx1250) wave32 WMMA cross-attention.
// D = A(16x32 f16) x B(32x16 f16) + C(16x16 f32) via v_wmma_f32_16x16x32_f16.

typedef __attribute__((ext_vector_type(16))) _Float16 v16h;
typedef __attribute__((ext_vector_type(8)))  float    v8f;

union Frag16 { v16h v; float4 q[2]; _Float16 h[16]; };

// ---- WMMA fragment loaders (CDNA5 ISA 7.12.2 layouts) -----------------------

// A-matrix 16x32 f16: lane L<16 holds row L with K {0..7,16..23};
// lane L>=16 holds row L-16 with K {8..15,24..31}. 16 halves per lane.
__device__ __forceinline__ v16h lds_load_afrag(const _Float16* base, int strideH) {
  const int lane = threadIdx.x & 31;
  const int row  = lane & 15;
  const int koff = (lane >> 4) << 3;              // 0 or 8
  const _Float16* p = base + row * strideH + koff;
  Frag16 f;
  f.q[0] = *(const float4*)(p);                   // K koff..koff+7
  f.q[1] = *(const float4*)(p + 16);              // K koff+16..koff+23
  return f.v;
}

// B-matrix 32x16 f16: lane L holds column L&15 with K {0..7,16..23} (L<16)
// or K {8..15,24..31} (L>=16).
__device__ __forceinline__ v16h lds_load_bfrag(const _Float16* base, int strideH) {
  const int lane = threadIdx.x & 31;
  const int koff = (lane >> 4) << 3;
  const _Float16* p = base + (lane & 15);
  Frag16 f;
#pragma unroll
  for (int s = 0; s < 8; ++s) {
    f.h[s]     = p[(koff + s) * strideH];
    f.h[s + 8] = p[(koff + 16 + s) * strideH];
  }
  return f.v;
}

__device__ __forceinline__ v8f wmma_f16(v16h a, v16h b, v8f c) {
  return __builtin_amdgcn_wmma_f32_16x16x32_f16(false, a, false, b, (short)0, c,
                                                false, false);
}

// ---- fused f32->f16 (or f16 copy) staging helpers ---------------------------

__device__ __forceinline__ void cvt16(_Float16* dst, const float* src) {
  float b[16];
  ((float4*)b)[0] = ((const float4*)src)[0];
  ((float4*)b)[1] = ((const float4*)src)[1];
  ((float4*)b)[2] = ((const float4*)src)[2];
  ((float4*)b)[3] = ((const float4*)src)[3];
  _Float16 h[16];
#pragma unroll
  for (int i = 0; i < 16; ++i) h[i] = (_Float16)b[i];
  ((float4*)dst)[0] = ((float4*)h)[0];
  ((float4*)dst)[1] = ((float4*)h)[1];
}
__device__ __forceinline__ void cvt16(_Float16* dst, const _Float16* src) {
  ((float4*)dst)[0] = ((const float4*)src)[0];
  ((float4*)dst)[1] = ((const float4*)src)[1];
}

// ---- GEMM: C[M,N] = A[M,K] @ W[N,K]^T + bias, epilogue scatters -------------

#define TILE_M 128
#define TILE_N 128
#define TILE_K 32
#define LDA_S  40    // padded f16 row stride for As (80B, 16B-aligned chunks)
#define LDB_S  136   // padded f16 row stride for Bs (272B)

enum { MODE_PROJ = 0, MODE_Q = 1, MODE_KV = 2 };

template <typename TA>
__global__ __launch_bounds__(256) void gemm_wmma_kernel(
    const TA* __restrict__ A, const float* __restrict__ W,
    const float* __restrict__ bias, float* __restrict__ outF,
    _Float16* __restrict__ outK, _Float16* __restrict__ outV, int M, int N,
    int K, int mode) {
  __shared__ _Float16 As[TILE_M * LDA_S];
  __shared__ _Float16 Bs[TILE_K * LDB_S];

  const int tid  = threadIdx.x;
  const int wave = tid >> 5;
  const int lane = tid & 31;
  const int m0g  = blockIdx.y * TILE_M;
  const int n0g  = blockIdx.x * TILE_N;
  const int wm   = (wave & 3) * 32;   // wave M offset in tile
  const int wn   = (wave >> 2) * 64;  // wave N offset in tile

  v8f acc[2][4];
#pragma unroll
  for (int i = 0; i < 2; ++i)
#pragma unroll
    for (int j = 0; j < 4; ++j) acc[i][j] = {};

  for (int k0 = 0; k0 < K; k0 += TILE_K) {
    __syncthreads();
    {  // stage A tile (convert to f16)
      const int row = tid >> 1, part = (tid & 1) << 4;
      cvt16(As + row * LDA_S + part, A + (size_t)(m0g + row) * K + k0 + part);
    }
    {  // stage W tile transposed into Bs[k][n]
      const int n = tid >> 1, part = (tid & 1) << 4;
      const float* src = W + (size_t)(n0g + n) * K + k0 + part;
      float b[16];
      ((float4*)b)[0] = ((const float4*)src)[0];
      ((float4*)b)[1] = ((const float4*)src)[1];
      ((float4*)b)[2] = ((const float4*)src)[2];
      ((float4*)b)[3] = ((const float4*)src)[3];
#pragma unroll
      for (int i = 0; i < 16; ++i) Bs[(part + i) * LDB_S + n] = (_Float16)b[i];
    }
    __syncthreads();

    v16h af[2], bf[4];
#pragma unroll
    for (int i = 0; i < 2; ++i)
      af[i] = lds_load_afrag(As + (wm + i * 16) * LDA_S, LDA_S);
#pragma unroll
    for (int j = 0; j < 4; ++j)
      bf[j] = lds_load_bfrag(Bs + wn + j * 16, LDB_S);
#pragma unroll
    for (int i = 0; i < 2; ++i)
#pragma unroll
      for (int j = 0; j < 4; ++j)
        acc[i][j] = wmma_f16(af[i], bf[j], acc[i][j]);
  }

  // epilogue: C/D layout — lane holds column (lane&15), rows (lane>>4)*8 + r
  const int col   = lane & 15;
  const int rbase = (lane >> 4) << 3;
#pragma unroll
  for (int i = 0; i < 2; ++i) {
#pragma unroll
    for (int j = 0; j < 4; ++j) {
      const int n  = n0g + wn + j * 16 + col;
      const float bv = bias[n];
#pragma unroll
      for (int r = 0; r < 8; ++r) {
        const int m = m0g + wm + i * 16 + rbase + r;
        const float val = acc[i][j][r] + bv;
        if (mode == MODE_PROJ) {
          outF[(size_t)m * N + n] = val;
        } else if (mode == MODE_Q) {
          // m = b*1024 + mt ; n = h*64 + d ; fold softmax scale 1/sqrt(64)
          const int b = m >> 10, mt = m & 1023, h = n >> 6, d = n & 63;
          outK[((((size_t)b * 16 + h) * 1024) + mt) * 64 + d] =
              (_Float16)(val * 0.125f);
        } else {  // MODE_KV: n<1024 -> K, else V ; m = b*2048 + ms
          const int b = m >> 11, ms = m & 2047;
          const int np = n & 1023, h = np >> 6, d = np & 63;
          _Float16* dst = (n < 1024) ? outK : outV;
          dst[((((size_t)b * 16 + h) * 2048) + ms) * 64 + d] = (_Float16)val;
        }
      }
    }
  }
}

// ---- flash attention: per (b,h), 64 q-rows per workgroup, 4 waves -----------

__global__ __launch_bounds__(128) void attn_wmma_kernel(
    const _Float16* __restrict__ Q, const _Float16* __restrict__ Kx,
    const _Float16* __restrict__ Vx, const unsigned char* __restrict__ mask,
    _Float16* __restrict__ O, int Nt, int Ns) {
  __shared__ _Float16 Qs[64 * 72];       // q tile   [row][dh]
  __shared__ _Float16 Kt[64 * 72];       // k chunk  [dh][key]  (transposed)
  __shared__ _Float16 Vs[64 * 64];       // v chunk  [key][dh]
  __shared__ _Float16 Ps[4][16 * 72];    // per-wave P transpose staging

  const int tid = threadIdx.x, wave = tid >> 5, lane = tid & 31;
  const int bh = blockIdx.y, b = bh >> 4, h = bh & 15;
  const int m0 = blockIdx.x * 64;

  const _Float16* Qg = Q + ((size_t)bh * Nt + m0) * 64;
  const _Float16* Kg = Kx + (size_t)bh * Ns * 64;
  const _Float16* Vg = Vx + (size_t)bh * Ns * 64;
  const unsigned char* mg = mask + (size_t)b * Ns;

  {  // stage 64x64 q tile
    const int row = tid >> 1, part = (tid & 1) << 5;
    const float4* sp = (const float4*)(Qg + (size_t)row * 64 + part);
    float4* dp = (float4*)(Qs + row * 72 + part);
    dp[0] = sp[0]; dp[1] = sp[1]; dp[2] = sp[2]; dp[3] = sp[3];
  }
  __syncthreads();

  const int wm = wave * 16;
  v16h qf[2];
  qf[0] = lds_load_afrag(Qs + wm * 72, 72);       // dh 0..31
  qf[1] = lds_load_afrag(Qs + wm * 72 + 32, 72);  // dh 32..63

  const int col = lane & 15, rbase = (lane >> 4) << 3;
  float mrow[8], lrow[8];
#pragma unroll
  for (int r = 0; r < 8; ++r) { mrow[r] = -1e30f; lrow[r] = 0.f; }
  v8f oacc[4];
#pragma unroll
  for (int j = 0; j < 4; ++j) oacc[j] = {};

  for (int kc = 0; kc < Ns; kc += 64) {
    __syncthreads();
    {  // stage K chunk transposed: Kt[dh][key]
      const int key = tid & 63, dh0 = (tid >> 6) << 5;
      const _Float16* sp = Kg + (size_t)(kc + key) * 64 + dh0;
      _Float16 hb[32];
      ((float4*)hb)[0] = ((const float4*)sp)[0];
      ((float4*)hb)[1] = ((const float4*)sp)[1];
      ((float4*)hb)[2] = ((const float4*)sp)[2];
      ((float4*)hb)[3] = ((const float4*)sp)[3];
#pragma unroll
      for (int i = 0; i < 32; ++i) Kt[(dh0 + i) * 72 + key] = hb[i];
    }
    {  // stage V chunk row-major
      const int row = tid >> 1, part = (tid & 1) << 5;
      const float4* sp = (const float4*)(Vg + (size_t)(kc + row) * 64 + part);
      float4* dp = (float4*)(Vs + row * 64 + part);
      dp[0] = sp[0]; dp[1] = sp[1]; dp[2] = sp[2]; dp[3] = sp[3];
    }
    __syncthreads();

    // S = Q @ K^T (q pre-scaled): 4 column fragments of 16 keys
    v8f s[4];
#pragma unroll
    for (int j = 0; j < 4; ++j) {
      s[j] = {};
      v16h b0 = lds_load_bfrag(Kt + 0 * 72 + j * 16, 72);
      v16h b1 = lds_load_bfrag(Kt + 32 * 72 + j * 16, 72);
      s[j] = wmma_f16(qf[0], b0, s[j]);
      s[j] = wmma_f16(qf[1], b1, s[j]);
    }

    // padding mask: True -> -1e30 (finite, avoids inf-inf NaN paths)
    float bm[4];
#pragma unroll
    for (int j = 0; j < 4; ++j) bm[j] = mg[kc + j * 16 + col] ? -1e30f : 0.f;
#pragma unroll
    for (int j = 0; j < 4; ++j)
#pragma unroll
      for (int r = 0; r < 8; ++r) s[j][r] += bm[j];

    // online softmax: row stats live redundantly in all 16 lanes of a half
    float corr[8];
#pragma unroll
    for (int r = 0; r < 8; ++r) {
      float v = fmaxf(fmaxf(s[0][r], s[1][r]), fmaxf(s[2][r], s[3][r]));
      v = fmaxf(v, __shfl_xor(v, 1, 32));
      v = fmaxf(v, __shfl_xor(v, 2, 32));
      v = fmaxf(v, __shfl_xor(v, 4, 32));
      v = fmaxf(v, __shfl_xor(v, 8, 32));
      const float mnew = fmaxf(mrow[r], v);
      corr[r] = __expf(mrow[r] - mnew);
      mrow[r] = mnew;
    }
    float rsum[8];
#pragma unroll
    for (int r = 0; r < 8; ++r) rsum[r] = 0.f;
#pragma unroll
    for (int j = 0; j < 4; ++j)
#pragma unroll
      for (int r = 0; r < 8; ++r) {
        const float p = __expf(s[j][r] - mrow[r]);
        s[j][r] = p;
        rsum[r] += p;
      }
#pragma unroll
    for (int r = 0; r < 8; ++r) {
      float v = rsum[r];
      v += __shfl_xor(v, 1, 32);
      v += __shfl_xor(v, 2, 32);
      v += __shfl_xor(v, 4, 32);
      v += __shfl_xor(v, 8, 32);
      lrow[r] = lrow[r] * corr[r] + v;
    }
#pragma unroll
    for (int j = 0; j < 4; ++j)
#pragma unroll
      for (int r = 0; r < 8; ++r) oacc[j][r] *= corr[r];

    // P: C-layout -> LDS -> A-layout (per-wave region; LDS is in-order per
    // wave, s_wait_dscnt 0 orders the reload after the scatter stores)
    _Float16* pp = Ps[wave];
#pragma unroll
    for (int j = 0; j < 4; ++j)
#pragma unroll
      for (int r = 0; r < 8; ++r)
        pp[(rbase + r) * 72 + j * 16 + col] = (_Float16)s[j][r];
    asm volatile("s_wait_dscnt 0" ::: "memory");

    v16h pf0 = lds_load_afrag(pp, 72);       // keys 0..31
    v16h pf1 = lds_load_afrag(pp + 32, 72);  // keys 32..63
#pragma unroll
    for (int j = 0; j < 4; ++j) {
      v16h vb0 = lds_load_bfrag(Vs + 0 * 64 + j * 16, 64);
      v16h vb1 = lds_load_bfrag(Vs + 32 * 64 + j * 16, 64);
      oacc[j] = wmma_f16(pf0, vb0, oacc[j]);
      oacc[j] = wmma_f16(pf1, vb1, oacc[j]);
    }
  }

  // normalize + store: O[b][m][h*64+d] (heads concatenated, f16)
  _Float16* Og = O + ((size_t)b * Nt + (m0 + wm)) * 1024 + h * 64;
#pragma unroll
  for (int j = 0; j < 4; ++j)
#pragma unroll
    for (int r = 0; r < 8; ++r) {
      const float val = oacc[j][r] / lrow[r];
      Og[(size_t)(rbase + r) * 1024 + j * 16 + col] = (_Float16)val;
    }
}

// ---- launch -----------------------------------------------------------------

extern "C" void kernel_launch(void* const* d_in, const int* in_sizes, int n_in,
                              void* d_out, int out_size, void* d_ws,
                              size_t ws_size, hipStream_t stream) {
  const float* tgt  = (const float*)d_in[0];
  const float* src  = (const float*)d_in[1];
  const unsigned char* pmask = (const unsigned char*)d_in[2];
  const float* q_w  = (const float*)d_in[3];
  const float* q_b  = (const float*)d_in[4];
  const float* kv_w = (const float*)d_in[5];
  const float* kv_b = (const float*)d_in[6];
  const float* p_w  = (const float*)d_in[7];
  const float* p_b  = (const float*)d_in[8];
  float* out = (float*)d_out;

  // workspace: Q16 8MB | K16 16MB | V16 16MB | O16 8MB  (all f16)
  _Float16* Q16 = (_Float16*)d_ws;
  _Float16* K16 = Q16 + (size_t)4 * 16 * 1024 * 64;
  _Float16* V16 = K16 + (size_t)4 * 16 * 2048 * 64;
  _Float16* O16 = V16 + (size_t)4 * 16 * 2048 * 64;

  const dim3 blk(256);
  // Q = tgt @ q_w^T + q_b, scaled, scattered to [B,H,Nt,Dh]
  gemm_wmma_kernel<float><<<dim3(1024 / 128, 4096 / 128), blk, 0, stream>>>(
      tgt, q_w, q_b, nullptr, Q16, nullptr, 4096, 1024, 1024, MODE_Q);
  // KV = src @ kv_w^T + kv_b, scattered to K/V [B,H,Ns,Dh]
  gemm_wmma_kernel<float><<<dim3(2048 / 128, 8192 / 128), blk, 0, stream>>>(
      src, kv_w, kv_b, nullptr, K16, V16, 8192, 2048, 1024, MODE_KV);
  // masked softmax attention -> O16 [B,Nt,DIM]
  attn_wmma_kernel<<<dim3(1024 / 64, 64), dim3(128), 0, stream>>>(
      Q16, K16, V16, pmask, O16, 1024, 2048);
  // out = O16 @ proj_w^T + proj_b (f32)
  gemm_wmma_kernel<_Float16><<<dim3(1024 / 128, 4096 / 128), blk, 0, stream>>>(
      O16, p_w, p_b, out, nullptr, nullptr, 4096, 1024, 1024, MODE_PROJ);
}
